// MoE_41016937677224
// MI455X (gfx1250) — matrix operand order repaired and checked by reading, not compiled, not run on hardware
//
#include <hip/hip_runtime.h>
#include <hip/hip_bf16.h>

typedef __attribute__((ext_vector_type(16))) __bf16 v16bf;
typedef __attribute__((ext_vector_type(8)))  __bf16 v8bf;
typedef __attribute__((ext_vector_type(2)))  __bf16 v2bf;
typedef __attribute__((ext_vector_type(8)))  float  v8f;

#define N_TOK 4096
#define DIM   768
#define NEXP  8
#define HDIM  7680
#define ODIM  768

#define TM 128
#define TN 128
#define KB 32
#define LSTR 40   // padded LDS row stride (bf16 elems): 80B = 20 banks -> conflict-free

__device__ __forceinline__ __bf16 f2bf(float f) { return (__bf16)f; }

__global__ void moe_zero_counts(int* counts) {
    if (threadIdx.x < NEXP) counts[threadIdx.x] = 0;
}

// One thread per token: logits over 8 experts, softmax top-1, append to expert list.
__global__ void moe_gate(const float* __restrict__ x,
                         const float* __restrict__ Wg,
                         const float* __restrict__ bg,
                         int* __restrict__ counts,
                         int* __restrict__ lists,
                         float* __restrict__ gate) {
    int n = blockIdx.x * blockDim.x + threadIdx.x;
    if (n >= N_TOK) return;
    float lg[NEXP];
    #pragma unroll
    for (int e = 0; e < NEXP; ++e) lg[e] = bg[e];
    const float* xr = x + (size_t)n * DIM;
    for (int d = 0; d < DIM; ++d) {
        float xv = xr[d];
        #pragma unroll
        for (int e = 0; e < NEXP; ++e) lg[e] += xv * Wg[d * NEXP + e];
    }
    int best = 0; float mx = lg[0];
    #pragma unroll
    for (int e = 1; e < NEXP; ++e) { if (lg[e] > mx) { mx = lg[e]; best = e; } }
    float s = 0.f;
    #pragma unroll
    for (int e = 0; e < NEXP; ++e) s += __expf(lg[e] - mx);
    gate[n] = 1.0f / s;                 // top prob = exp(0)/sum
    int pos = atomicAdd(&counts[best], 1);
    lists[best * N_TOK + pos] = n;
}

// Tiled bf16 WMMA GEMM over one expert's gathered tokens.
// LAYER 1: A = gather(x) fp32->bf16, B = W1[e], epilogue relu -> bf16 h
// LAYER 2: A = gather(h) bf16,       B = W2[e], epilogue (+b2)*gate -> fp32 out
template<int LAYER>
__global__ __launch_bounds__(256) void moe_gemm(
    const float*  __restrict__ x,
    const __bf16* __restrict__ hin,
    const float*  __restrict__ W,
    const float*  __restrict__ bias,
    const int*    __restrict__ counts,
    const int*    __restrict__ lists,
    const float*  __restrict__ gate,
    __bf16*       __restrict__ hout,
    float*        __restrict__ out)
{
    constexpr int KDIM = (LAYER == 1) ? DIM  : HDIM;
    constexpr int NCOL = (LAYER == 1) ? HDIM : ODIM;

    const int e      = blockIdx.z;
    const int tile_m = blockIdx.y;
    const int tile_n = blockIdx.x;
    const int count  = counts[e];
    if (tile_m * TM >= count) return;   // block-uniform: WMMA keeps full EXEC

    __shared__ alignas(16) __bf16 sA[TM * LSTR];
    __shared__ alignas(16) __bf16 sB[TN * LSTR];

    const int t    = threadIdx.x;
    const int lane = t & 31;
    const int wave = t >> 5;
    const int wm   = wave & 3;   // 4 M-blocks of 32 rows
    const int wn   = wave >> 2;  // 2 N-blocks of 64 cols
    const int lm   = lane & 15;
    const int lh   = lane >> 4;

    const int*   list = lists + e * N_TOK;
    const float* wB   = W + (size_t)e * KDIM * NCOL;

    // A staging: 2 threads per row, 16 contiguous K elems each
    const int sa_row = t >> 1;
    const int sa_kh  = (t & 1) << 4;
    int a_idx = tile_m * TM + sa_row;
    if (a_idx > count - 1) a_idx = count - 1;     // clamp (count >= 1 here)
    const int a_tok = list[a_idx];
    // B staging: thread owns k-row (t>>3), 16 contiguous n starting (t&7)*16
    const int sb_k  = t >> 3;
    const int sb_nh = (t & 7) << 4;

    const v8f vzero = {0.f,0.f,0.f,0.f,0.f,0.f,0.f,0.f};
    v8f acc[2][4];
    #pragma unroll
    for (int i = 0; i < 2; ++i)
        #pragma unroll
        for (int j = 0; j < 4; ++j) acc[i][j] = vzero;

    for (int kb = 0; kb < KDIM; kb += KB) {
        __syncthreads();
        // ---- stage A tile (row-major [TM][KB]) ----
        if (LAYER == 1) {
            const float4* src = (const float4*)(x + (size_t)a_tok * KDIM + kb + sa_kh);
            #pragma unroll
            for (int j = 0; j < 4; ++j) {
                float4 v = src[j];
                v2bf p0; p0[0] = f2bf(v.x); p0[1] = f2bf(v.y);
                v2bf p1; p1[0] = f2bf(v.z); p1[1] = f2bf(v.w);
                *(v2bf*)&sA[sa_row * LSTR + sa_kh + j * 4]     = p0;
                *(v2bf*)&sA[sa_row * LSTR + sa_kh + j * 4 + 2] = p1;
            }
        } else {
            const v8bf* src = (const v8bf*)(hin + (size_t)a_tok * KDIM + kb + sa_kh);
            *(v8bf*)&sA[sa_row * LSTR + sa_kh]     = src[0];
            *(v8bf*)&sA[sa_row * LSTR + sa_kh + 8] = src[1];
        }
        // ---- stage B tile transposed into LDS: sB[n][k] ----
        {
            const float4* src = (const float4*)(wB + (size_t)(kb + sb_k) * NCOL
                                                + tile_n * TN + sb_nh);
            #pragma unroll
            for (int j = 0; j < 4; ++j) {
                float4 v = src[j];
                sB[(sb_nh + j * 4 + 0) * LSTR + sb_k] = f2bf(v.x);
                sB[(sb_nh + j * 4 + 1) * LSTR + sb_k] = f2bf(v.y);
                sB[(sb_nh + j * 4 + 2) * LSTR + sb_k] = f2bf(v.z);
                sB[(sb_nh + j * 4 + 3) * LSTR + sb_k] = f2bf(v.w);
            }
        }
        __syncthreads();

        // ---- fragment loads (k 8*lh..+7 and 16+8*lh..+7 per lane, per ISA 16-bit layouts) ----
        v16bf a[2], b[4];
        #pragma unroll
        for (int mb = 0; mb < 2; ++mb) {
            int row = wm * 32 + mb * 16 + lm;
            v8bf lo = *(const v8bf*)&sA[row * LSTR + (lh << 3)];        // K = base..base+7
            v8bf hi = *(const v8bf*)&sA[row * LSTR + 16 + (lh << 3)];   // K = 16+base..
            a[mb] = __builtin_shufflevector(lo, hi, 0,1,2,3,4,5,6,7,8,9,10,11,12,13,14,15);
        }
        #pragma unroll
        for (int nb = 0; nb < 4; ++nb) {
            int col = wn * 64 + nb * 16 + lm;
            v8bf lo = *(const v8bf*)&sB[col * LSTR + (lh << 3)];        // K = base..base+7
            v8bf hi = *(const v8bf*)&sB[col * LSTR + 16 + (lh << 3)];   // K = 16+base..
            b[nb] = __builtin_shufflevector(lo, hi, 0,1,2,3,4,5,6,7,8,9,10,11,12,13,14,15);
        }
        #pragma unroll
        for (int mb = 0; mb < 2; ++mb)
            #pragma unroll
            for (int nb = 0; nb < 4; ++nb)
                acc[mb][nb] = __builtin_amdgcn_wmma_f32_16x16x32_bf16(
                    false, a[mb], false, b[nb], (short)0, acc[mb][nb], false, false);
    }

    // ---- epilogue (C/D layout: VGPR r -> row r + 8*lh, col = lm) ----
    #pragma unroll
    for (int mb = 0; mb < 2; ++mb) {
        #pragma unroll
        for (int r = 0; r < 8; ++r) {
            int i_local = wm * 32 + mb * 16 + r + lh * 8;
            int idx = tile_m * TM + i_local;
            if (idx >= count) continue;
            int tok = list[idx];
            #pragma unroll
            for (int nb = 0; nb < 4; ++nb) {
                int col = tile_n * TN + wn * 64 + nb * 16 + lm;
                float v = acc[mb][nb][r] + bias[e * NCOL + col];
                if (LAYER == 1) {
                    v = v > 0.f ? v : 0.f;
                    hout[(size_t)tok * HDIM + col] = f2bf(v);
                } else {
                    out[(size_t)tok * ODIM + col] = v * gate[tok];
                }
            }
        }
    }
}

extern "C" void kernel_launch(void* const* d_in, const int* in_sizes, int n_in,
                              void* d_out, int out_size, void* d_ws, size_t ws_size,
                              hipStream_t stream) {
    const float* x  = (const float*)d_in[0];
    const float* Wg = (const float*)d_in[1];
    const float* bg = (const float*)d_in[2];
    const float* W1 = (const float*)d_in[3];
    const float* b1 = (const float*)d_in[4];
    const float* W2 = (const float*)d_in[5];
    const float* b2 = (const float*)d_in[6];
    float* out = (float*)d_out;

    // workspace layout (all offsets 256B aligned):
    //   [0,32)          int counts[8]
    //   [256,131328)    int lists[E][N]
    //   [131328,147712) float gate[N]
    //   [147712,+63MB)  bf16 h[N][HDIM]
    char* ws = (char*)d_ws;
    int*    counts = (int*)ws;
    int*    lists  = (int*)(ws + 256);
    float*  gatev  = (float*)(ws + 131328);
    __bf16* hb     = (__bf16*)(ws + 147712);

    moe_zero_counts<<<1, 32, 0, stream>>>(counts);
    moe_gate<<<(N_TOK + 255) / 256, 256, 0, stream>>>(x, Wg, bg, counts, lists, gatev);

    dim3 g1(HDIM / TN, N_TOK / TM, NEXP);   // 60 x 32 x 8, most tiles early-exit
    moe_gemm<1><<<g1, 256, 0, stream>>>(x, hb, W1, b1, counts, lists, gatev, hb, out);

    dim3 g2(ODIM / TN, N_TOK / TM, NEXP);   // 6 x 32 x 8
    moe_gemm<2><<<g2, 256, 0, stream>>>(x, hb, W2, b2, counts, lists, gatev, hb, out);
}
